// RGCNForGraphClassification_56667798503745
// MI455X (gfx1250) — matrix operand order repaired and checked by reading, not compiled, not run on hardware
//
#include <hip/hip_runtime.h>
#include <hip/hip_bf16.h>

// ---------------------------------------------------------------------------
// RGCN (basis decomposition) graph classification, CDNA5 / gfx1250.
// Dense transforms via v_wmma_f32_16x16x32_f16 (wave32), sparse aggregation
// via scaled f32 atomics (mean folded into per-edge scale).
// GEMM: wave computes 32(M) x 64(N); 4 B fragments shared across two
// M-subtiles; u32 saddr-form addressing (3 offset VGPRs total), branch-free
// inner loop.
// ---------------------------------------------------------------------------

#define N_NODES 50000
#define N_EDGES 600000
#define IN_CH   256
#define HID     128
#define OUT_CH  8
#define N_REL   4
#define N_BASES 8
#define N_GRAPHS 64
#define NCAT    (N_REL * HID + HID)   // 640: 4 relation blocks + root block

typedef __attribute__((ext_vector_type(16))) _Float16 v16h;
typedef __attribute__((ext_vector_type(8)))  _Float16 v8h;
typedef __attribute__((ext_vector_type(8)))  float    v8f;

// ---------------- helpers: zero / convert ----------------------------------

__global__ void k_zero_f32(float* __restrict__ p, int n) {
    int i = blockIdx.x * blockDim.x + threadIdx.x;
    if (i < n) p[i] = 0.0f;
}

__global__ void k_f32_to_f16(const float* __restrict__ in, _Float16* __restrict__ out, int n) {
    int i = blockIdx.x * blockDim.x + threadIdx.x;
    if (i < n) out[i] = (_Float16)in[i];
}

// ---------------- weight prep: Wt[n][k] = cat(relations, root), N-major -----

__global__ void k_prep_w(const float* __restrict__ bases,  // [B][K][HID]
                         const float* __restrict__ comp,   // [R][B]
                         const float* __restrict__ root,   // [K][HID]
                         _Float16* __restrict__ wt,        // [NCAT][K]
                         int K) {
    int i = blockIdx.x * blockDim.x + threadIdx.x;
    if (i >= NCAT * K) return;
    int n = i / K;           // output column 0..639
    int k = i % K;           // input channel
    float acc;
    if (n < N_REL * HID) {
        int r = n >> 7;          // n / 128
        int c = n & 127;         // n % 128
        acc = 0.0f;
        #pragma unroll
        for (int b = 0; b < N_BASES; ++b)
            acc += comp[r * N_BASES + b] * bases[(size_t)b * K * HID + (size_t)k * HID + c];
    } else {
        int c = n - N_REL * HID;
        acc = root[(size_t)k * HID + c];
    }
    wt[(size_t)n * K + k] = (_Float16)acc;
}

// ---------------- WMMA GEMM: C[M][NCAT] = A[M][K] * Wt^T --------------------
// A row-major f16, Wt is [NCAT][K] f16 (each output column contiguous in K).
// block = 128 threads = 4 waves; each wave -> 32 rows x 64 cols (two 16-row
// subtiles sharing the 4 B fragments). M multiple of 16. K compile-time.
// All loads: scalar base + u32 lane offset; B streams use immediate deltas.

template <int K>
__global__ void k_gemm_wmma(const _Float16* __restrict__ A,
                            const _Float16* __restrict__ Wt,
                            float* __restrict__ C,
                            int M) {
    const int lane    = threadIdx.x & 31;
    const int wave    = threadIdx.x >> 5;
    const int halfSel = lane >> 4;       // 0: lanes 0-15, 1: lanes 16-31
    const int l16     = lane & 15;

    const int m0 = blockIdx.x * 128 + wave * 32;   // first 16-row subtile
    if (m0 + 16 > M) return;                       // uniform per wave
    const bool run1 = (m0 + 32 <= M);              // uniform per wave
    const int m1 = run1 ? (m0 + 16) : m0;          // clamp (redundant compute;
                                                   // store guarded below)
    const int n0 = blockIdx.y * 64;

    const char* Ab = (const char*)A;
    const char* Wb = (const char*)Wt;

    // u32 byte offsets (all buffers < 4GB). One offset per A subtile; the 4
    // B column-groups are at compile-time deltas of 16*K*2 bytes.
    // A frag: K-index kk + halfSel*16 + {0..7} (lo) and +8 more (hi).
    uint32_t aoff0 = ((uint32_t)(m0 + l16) * K + (uint32_t)halfSel * 16) * 2u;
    uint32_t aoff1 = ((uint32_t)(m1 + l16) * K + (uint32_t)halfSel * 16) * 2u;
    // B frag: K-index kk + halfSel*16 + {0..15} (one 32B load).
    uint32_t boff  = ((uint32_t)(n0 + l16) * K + (uint32_t)halfSel * 16) * 2u;
    constexpr uint32_t BSTEP = 16u * K * 2u;       // bytes between column groups

    v8f acc00 = {}, acc01 = {}, acc02 = {}, acc03 = {};
    v8f acc10 = {}, acc11 = {}, acc12 = {}, acc13 = {};

    for (uint32_t kb = 0; kb < (uint32_t)(K * 2); kb += 64) {
        v8h lo0 = *(const v8h*)(Ab + (aoff0 + kb));
        v8h hi0 = *(const v8h*)(Ab + (aoff0 + kb) + 16);
        v16h a0 = __builtin_shufflevector(lo0, hi0,
                  0,1,2,3,4,5,6,7,8,9,10,11,12,13,14,15);
        v8h lo1 = *(const v8h*)(Ab + (aoff1 + kb));
        v8h hi1 = *(const v8h*)(Ab + (aoff1 + kb) + 16);
        v16h a1 = __builtin_shufflevector(lo1, hi1,
                  0,1,2,3,4,5,6,7,8,9,10,11,12,13,14,15);

        v16h b0 = *(const v16h*)(Wb + (boff + kb));
        v16h b1 = *(const v16h*)(Wb + (boff + kb) + BSTEP);
        v16h b2 = *(const v16h*)(Wb + (boff + kb) + 2 * BSTEP);
        v16h b3 = *(const v16h*)(Wb + (boff + kb) + 3 * BSTEP);

        acc00 = __builtin_amdgcn_wmma_f32_16x16x32_f16(false, a0, false, b0, (short)0, acc00, false, false);
        acc01 = __builtin_amdgcn_wmma_f32_16x16x32_f16(false, a0, false, b1, (short)0, acc01, false, false);
        acc02 = __builtin_amdgcn_wmma_f32_16x16x32_f16(false, a0, false, b2, (short)0, acc02, false, false);
        acc03 = __builtin_amdgcn_wmma_f32_16x16x32_f16(false, a0, false, b3, (short)0, acc03, false, false);
        acc10 = __builtin_amdgcn_wmma_f32_16x16x32_f16(false, a1, false, b0, (short)0, acc10, false, false);
        acc11 = __builtin_amdgcn_wmma_f32_16x16x32_f16(false, a1, false, b1, (short)0, acc11, false, false);
        acc12 = __builtin_amdgcn_wmma_f32_16x16x32_f16(false, a1, false, b2, (short)0, acc12, false, false);
        acc13 = __builtin_amdgcn_wmma_f32_16x16x32_f16(false, a1, false, b3, (short)0, acc13, false, false);
    }

    // C/D layout: VGPR v, lane -> M = mbase + v + halfSel*8, N = n0 + t*16 + l16
    #pragma unroll
    for (int v = 0; v < 8; ++v) {
        float* crow = C + (size_t)(m0 + v + halfSel * 8) * NCAT + n0 + l16;
        crow[0]  = acc00[v];
        crow[16] = acc01[v];
        crow[32] = acc02[v];
        crow[48] = acc03[v];
    }
    if (run1) {
        #pragma unroll
        for (int v = 0; v < 8; ++v) {
            float* crow = C + (size_t)(m1 + v + halfSel * 8) * NCAT + n0 + l16;
            crow[0]  = acc10[v];
            crow[16] = acc11[v];
            crow[32] = acc12[v];
            crow[48] = acc13[v];
        }
    }
}

// ---------------- edge degree counts (shared by both layers) ----------------

__global__ void k_edge_count(const int* __restrict__ dst,
                             const int* __restrict__ etype,
                             float* __restrict__ cnt) {
    int e = blockIdx.x * blockDim.x + threadIdx.x;
    if (e < N_EDGES) atomicAdd(&cnt[(size_t)dst[e] * N_REL + etype[e]], 1.0f);
}

// ---------------- per-edge scaled scatter-add (mean aggregation) ------------
// One wave per edge: 32 lanes x float4 = 128 features.

__global__ void k_edge_aggregate(const int* __restrict__ src,
                                 const int* __restrict__ dst,
                                 const int* __restrict__ etype,
                                 const float* __restrict__ cnt,
                                 const float* __restrict__ hcat,  // [N][NCAT]
                                 float* __restrict__ agg) {       // [N][HID]
    const int lane = threadIdx.x & 31;
    const int e = blockIdx.x * (blockDim.x >> 5) + (threadIdx.x >> 5);
    if (e >= N_EDGES) return;
    const int r = etype[e];
    const int s = src[e];
    const int d = dst[e];
    const float c = cnt[(size_t)d * N_REL + r];
    const float scale = 1.0f / fmaxf(c, 1.0f);
    const float4 m = ((const float4*)(hcat + (size_t)s * NCAT + (size_t)r * HID))[lane];
    float* arow = agg + (size_t)d * HID + lane * 4;
    atomicAdd(&arow[0], m.x * scale);
    atomicAdd(&arow[1], m.y * scale);
    atomicAdd(&arow[2], m.z * scale);
    atomicAdd(&arow[3], m.w * scale);
}

// ---------------- combine: agg + root-part + bias, ReLU ---------------------

__global__ void k_combine_f16(const float* __restrict__ agg,
                              const float* __restrict__ hcat,
                              const float* __restrict__ bias,
                              _Float16* __restrict__ out) {      // layer-1 -> f16 input of layer-2
    int i = blockIdx.x * blockDim.x + threadIdx.x;
    if (i >= N_NODES * HID) return;
    int n = i >> 7, c = i & 127;
    float v = agg[i] + hcat[(size_t)n * NCAT + N_REL * HID + c] + bias[c];
    out[i] = (_Float16)fmaxf(v, 0.0f);
}

__global__ void k_combine_f32(const float* __restrict__ agg,
                              const float* __restrict__ hcat,
                              const float* __restrict__ bias,
                              float* __restrict__ out) {         // layer-2 -> f32 for pooling
    int i = blockIdx.x * blockDim.x + threadIdx.x;
    if (i >= N_NODES * HID) return;
    int n = i >> 7, c = i & 127;
    float v = agg[i] + hcat[(size_t)n * NCAT + N_REL * HID + c] + bias[c];
    out[i] = fmaxf(v, 0.0f);
}

// ---------------- global mean pool ------------------------------------------

__global__ void k_pool(const float* __restrict__ h2,
                       const int* __restrict__ batch,
                       float* __restrict__ pooled,   // [G][HID]
                       float* __restrict__ gcnt) {   // [G]
    const int lane = threadIdx.x & 31;
    const int n = blockIdx.x * (blockDim.x >> 5) + (threadIdx.x >> 5);
    if (n >= N_NODES) return;
    const int g = batch[n];
    const float4 v = ((const float4*)(h2 + (size_t)n * HID))[lane];
    float* prow = pooled + (size_t)g * HID + lane * 4;
    atomicAdd(&prow[0], v.x);
    atomicAdd(&prow[1], v.y);
    atomicAdd(&prow[2], v.z);
    atomicAdd(&prow[3], v.w);
    if (lane == 0) atomicAdd(&gcnt[g], 1.0f);
}

// ---------------- head: (pooled/cnt) @ Wc + bc ------------------------------

__global__ void k_head(const float* __restrict__ pooled,
                       const float* __restrict__ gcnt,
                       const float* __restrict__ Wc,   // [HID][OUT_CH]
                       const float* __restrict__ bc,
                       float* __restrict__ out) {      // [G][OUT_CH]
    int i = blockIdx.x * blockDim.x + threadIdx.x;
    if (i >= N_GRAPHS * OUT_CH) return;
    int g = i / OUT_CH, o = i % OUT_CH;
    float inv = 1.0f / fmaxf(gcnt[g], 1.0f);
    float acc = 0.0f;
    for (int k = 0; k < HID; ++k)
        acc += pooled[(size_t)g * HID + k] * Wc[(size_t)k * OUT_CH + o];
    out[i] = acc * inv + bc[o];
}

// ---------------------------------------------------------------------------

extern "C" void kernel_launch(void* const* d_in, const int* in_sizes, int n_in,
                              void* d_out, int out_size, void* d_ws, size_t ws_size,
                              hipStream_t stream) {
    (void)in_sizes; (void)n_in; (void)out_size; (void)ws_size;

    const float* x       = (const float*)d_in[0];     // [N, 256]
    const int*   eidx    = (const int*)d_in[1];       // [2, E]
    const int*   etype   = (const int*)d_in[2];       // [E]
    const int*   batch   = (const int*)d_in[3];       // [N]
    const float* bases1  = (const float*)d_in[4];
    const float* comp1   = (const float*)d_in[5];
    const float* root1   = (const float*)d_in[6];
    const float* bias1   = (const float*)d_in[7];
    const float* bases2  = (const float*)d_in[8];
    const float* comp2   = (const float*)d_in[9];
    const float* root2   = (const float*)d_in[10];
    const float* bias2   = (const float*)d_in[11];
    const float* Wc      = (const float*)d_in[12];
    const float* bc      = (const float*)d_in[13];
    float* out           = (float*)d_out;

    const int* src = eidx;
    const int* dst = eidx + N_EDGES;

    // ---- workspace carve-up (256-byte aligned) ----
    char* ws = (char*)d_ws;
    size_t off = 0;
    auto carve = [&](size_t bytes) -> char* {
        char* p = ws + off;
        off += (bytes + 255) & ~(size_t)255;
        return p;
    };
    _Float16* xh     = (_Float16*)carve((size_t)N_NODES * IN_CH * 2);   // 25.6 MB
    _Float16* wt1    = (_Float16*)carve((size_t)NCAT * IN_CH * 2);      // 320 KB
    _Float16* wt2    = (_Float16*)carve((size_t)NCAT * HID * 2);        // 160 KB
    float*    hcat   = (float*)   carve((size_t)N_NODES * NCAT * 4);    // 128 MB (reused both layers)
    float*    cnt    = (float*)   carve((size_t)N_NODES * N_REL * 4);   // 800 KB
    float*    agg    = (float*)   carve((size_t)N_NODES * HID * 4);     // 25.6 MB (reused)
    _Float16* h1h    = (_Float16*)carve((size_t)N_NODES * HID * 2);     // 12.8 MB
    float*    h2     = (float*)   carve((size_t)N_NODES * HID * 4);     // 25.6 MB
    float*    pooled = (float*)   carve((size_t)N_GRAPHS * HID * 4);
    float*    gcnt   = (float*)   carve((size_t)N_GRAPHS * 4);

    const int TPB = 256;
    const int nAgg = N_NODES * HID;                       // 6.4M
    dim3 gemmGrid((N_NODES + 127) / 128, NCAT / 64);      // (391, 10); 128 rows/block
    const int edgeBlocks = (N_EDGES + 7) / 8;             // 8 waves/block, 1 edge/wave
    const int nodeBlocks = (N_NODES + 7) / 8;

    // 0) input prep
    k_f32_to_f16<<<(N_NODES * IN_CH + TPB - 1) / TPB, TPB, 0, stream>>>(x, xh, N_NODES * IN_CH);
    k_prep_w<<<(NCAT * IN_CH + TPB - 1) / TPB, TPB, 0, stream>>>(bases1, comp1, root1, wt1, IN_CH);
    k_prep_w<<<(NCAT * HID + TPB - 1) / TPB, TPB, 0, stream>>>(bases2, comp2, root2, wt2, HID);
    k_zero_f32<<<(N_NODES * N_REL + TPB - 1) / TPB, TPB, 0, stream>>>(cnt, N_NODES * N_REL);
    k_edge_count<<<(N_EDGES + TPB - 1) / TPB, TPB, 0, stream>>>(dst, etype, cnt);

    // 1) layer 1: dense WMMA transform -> scatter mean -> combine+ReLU -> f16
    k_gemm_wmma<IN_CH><<<gemmGrid, 128, 0, stream>>>(xh, wt1, hcat, N_NODES);
    k_zero_f32<<<(nAgg + TPB - 1) / TPB, TPB, 0, stream>>>(agg, nAgg);
    k_edge_aggregate<<<edgeBlocks, TPB, 0, stream>>>(src, dst, etype, cnt, hcat, agg);
    k_combine_f16<<<(nAgg + TPB - 1) / TPB, TPB, 0, stream>>>(agg, hcat, bias1, h1h);

    // 2) layer 2
    k_gemm_wmma<HID><<<gemmGrid, 128, 0, stream>>>(h1h, wt2, hcat, N_NODES);
    k_zero_f32<<<(nAgg + TPB - 1) / TPB, TPB, 0, stream>>>(agg, nAgg);
    k_edge_aggregate<<<edgeBlocks, TPB, 0, stream>>>(src, dst, etype, cnt, hcat, agg);
    k_combine_f32<<<(nAgg + TPB - 1) / TPB, TPB, 0, stream>>>(agg, hcat, bias2, h2);

    // 3) global mean pool + linear head
    k_zero_f32<<<(N_GRAPHS * HID + TPB - 1) / TPB, TPB, 0, stream>>>(pooled, N_GRAPHS * HID);
    k_zero_f32<<<1, N_GRAPHS, 0, stream>>>(gcnt, N_GRAPHS);
    k_pool<<<nodeBlocks, TPB, 0, stream>>>(h2, batch, pooled, gcnt);
    k_head<<<(N_GRAPHS * OUT_CH + TPB - 1) / TPB, TPB, 0, stream>>>(pooled, gcnt, Wc, bc, out);
}
